// EncoderRNN_79070347920104
// MI455X (gfx1250) — matrix operand order repaired and checked by reading, not compiled, not run on hardware
//
#include <hip/hip_runtime.h>

typedef __attribute__((ext_vector_type(16))) _Float16 v16h_t;
typedef __attribute__((ext_vector_type(8)))  _Float16 v8h_t;
typedef __attribute__((ext_vector_type(8)))  float    v8f_t;

#define GRU_H 22
#define GRU_T 1000
#define GRU_B 256
#define NG    66        // 3*H gates per direction
#define NG2   132       // both directions stacked
#define KP0   32        // padded K for layer-0 input (22 -> 32)
#define KP1   64        // padded K for layer-1 input (44 -> 64)
#define KLIN  22000
#define KPLIN 22016     // 22000 -> multiple of 32
#define GEMM_WAVES 4
#define GATE_U 11       // (16*22)/32 gate-math iterations per lane

// ---- CDNA5 WMMA f16 16x16x32 register layouts (CDNA5 ISA 7.12.2) ----
// A (16x32, MxK): lane m = lane&15; half-wave h = lane>>4.
//   a[0..7]  = K = 8h .. 8h+7     (V0..V3)
//   a[8..15] = K = 16+8h .. 23+8h (V4..V7)
//   => two contiguous 8-half (16B) loads per lane from a K-major f16 row.
// B (32x16, KxN): lane n = lane&15; b-element e=2v+lo holds
//   K = 2*(v&3) + 8*(lane>>4) + 16*(v>>2) + lo  -> packed offline into 16 contiguous halves/lane.
// D/C (16x16 f32): lane n = lane&15, M = vgpr + 8*(lane>>4)
__device__ __forceinline__ int b_kbase(int v, int half) {
    return ((v & 3) << 1) + (half << 3) + ((v >> 2) << 4);
}

// ---------------------------------------------------------------------------
// xt[(t*B + b)*32 + h] = f16( x[b, 0, h, t] ), zero for h >= 22.
__global__ void transpose_pad_kernel(const float* __restrict__ x,
                                     _Float16* __restrict__ xt) {
    long i = (long)blockIdx.x * blockDim.x + threadIdx.x;
    const long total = (long)GRU_T * GRU_B * KP0;
    if (i >= total) return;
    int  h  = (int)(i & (KP0 - 1));
    long tb = i >> 5;
    int  b  = (int)(tb % GRU_B);
    int  t  = (int)(tb / GRU_B);
    float v = (h < GRU_H) ? x[(long)b * GRU_H * GRU_T + (long)h * GRU_T + t] : 0.f;
    xt[i] = (_Float16)v;
}

// ---------------------------------------------------------------------------
// Pack W[N,K] (f32 row-major) into WMMA B-fragment layout:
// out[((nt*KC + kc)*32 + lane)*16 + e], e=2v+lo -> f16(W[col, kc*32 + bk + lo])
__global__ void pack_w_kernel(const float* __restrict__ W, int N, int K,
                              int KC, _Float16* __restrict__ out, long total) {
    long i = (long)blockIdx.x * blockDim.x + threadIdx.x;
    if (i >= total) return;
    int  e    = (int)(i & 15);
    int  lane = (int)((i >> 4) & 31);
    long ntkc = i >> 9;
    int  kc   = (int)(ntkc % KC);
    int  nt   = (int)(ntkc / KC);
    int  col  = nt * 16 + (lane & 15);
    int  k    = kc * 32 + b_kbase(e >> 1, lane >> 4) + (e & 1);
    out[i] = (_Float16)((col < N && k < K) ? W[(long)col * K + k] : 0.f);
}

// ---------------------------------------------------------------------------
// C[M,N] (f32, ldc) = relu?( A * W^T + bias ).
// A: f16, rows zero-padded to ldaPad. Wp: pre-packed B fragments.
// Inner loop: 2 x 16B A loads + 1 x 32B B load + 1 WMMA. No guards.
__global__ void wmma_gemm_f16(const _Float16* __restrict__ A, int ldaPad, int M,
                              int KC, const _Float16* __restrict__ Wp,
                              const float* __restrict__ bias, int N,
                              float* __restrict__ C, int ldc, int relu)
{
    int lane  = threadIdx.x & 31;
    int wave  = threadIdx.x >> 5;
    int mtile = blockIdx.x * GEMM_WAVES + wave;
    int ntile = blockIdx.y;
    if (mtile * 16 >= M) return;

    int half = lane >> 4;
    int lm   = lane & 15;
    int row  = mtile * 16 + lm;
    int col  = ntile * 16 + lm;

    const _Float16* arow = A + (long)row * ldaPad + 8 * half;
    const v16h_t*   bp   = (const v16h_t*)Wp + (long)ntile * KC * 32 + lane;

    v8f_t c = {};
    for (int kc = 0; kc < KC; ++kc) {
        v8h_t lo = *(const v8h_t*)(arow + kc * 32);
        v8h_t hi = *(const v8h_t*)(arow + kc * 32 + 16);
        v16h_t a;
#pragma unroll
        for (int e = 0; e < 8; ++e) { a[e] = lo[e]; a[8 + e] = hi[e]; }
        v16h_t b = bp[(long)kc * 32];
        c = __builtin_amdgcn_wmma_f32_16x16x32_f16(false, a, false, b,
                                                   (short)0, c, false, false);
    }

    float bv = (col < N) ? bias[col] : 0.f;
#pragma unroll
    for (int v = 0; v < 8; ++v) {
        int m = mtile * 16 + v + half * 8;
        if (col < N) {
            float val = c[v] + bv;
            if (relu) val = fmaxf(val, 0.f);
            C[(long)m * ldc + col] = val;
        }
    }
}

// ---------------------------------------------------------------------------
// One wave per (direction, 16-batch tile). w_hh lives in 5 resident v16h
// B-fragments. h kept twice in LDS: f32 (carry precision) + f16 padded to 32
// (A-fragment staging). Per step, the 33 xp loads per lane are issued FIRST
// (into registers) so their latency overlaps the A-build, 5 WMMAs and gh
// scatter; next step's xp rows are prefetched (global_prefetch_b8).
__global__ void gru_scan_kernel(const float*    __restrict__ xp,   // [T, B, 132] f32
                                const _Float16* __restrict__ whp,  // packed, [2][5][32] v16h
                                const float*    __restrict__ b_hh, // [2, 66]
                                _Float16*       __restrict__ hout) // [T*B, 64] f16 (pad pre-zeroed)
{
    __shared__ __align__(16) float    h_lds[16 * 24];   // f32 h, stride 24
    __shared__ __align__(16) _Float16 ha_lds[16 * 32];  // f16 h, stride 32, zero pad
    __shared__ __align__(16) float    gh_lds[16 * 80];  // gh rows (66 valid of 80)

    int lane = threadIdx.x;
    int half = lane >> 4;
    int lm   = lane & 15;
    int b0   = blockIdx.x * 16;
    int dir  = blockIdx.y;

    for (int i = lane; i < 16 * 24; i += 32) h_lds[i] = 0.f;
    for (int i = lane; i < 16 * 32; i += 32) ha_lds[i] = (_Float16)0.f;

    // resident recurrent-weight fragments + per-column bias
    v16h_t bw[5];
    float  bb[5];
    const v16h_t* wp = (const v16h_t*)whp + (long)dir * 5 * 32 + lane;
#pragma unroll
    for (int j = 0; j < 5; ++j) {
        bw[j] = wp[(long)j * 32];
        int colj = j * 16 + lm;
        bb[j] = (colj < NG) ? b_hh[dir * NG + colj] : 0.f;
    }

    // per-lane gate-slot indices (u-th slot: i = lane + 32u -> row m, hidden hx)
    int gm[GATE_U], gx[GATE_U];
#pragma unroll
    for (int u = 0; u < GATE_U; ++u) {
        int i = lane + u * 32;
        gm[u] = i / GRU_H;
        gx[u] = i - gm[u] * GRU_H;
    }
    __syncthreads();

    for (int t = 0; t < GRU_T; ++t) {
        int tt = dir ? (GRU_T - 1 - t) : t;
        const float* xbase = xp + ((long)tt * GRU_B + b0) * NG2 + dir * NG;

        // (1) issue this step's xp loads early: latency hides under WMMA phase
        float ir[GATE_U], iz[GATE_U], inn[GATE_U];
#pragma unroll
        for (int u = 0; u < GATE_U; ++u) {
            const float* xr = xbase + gm[u] * NG2;
            ir[u]  = xr[gx[u]];
            iz[u]  = xr[GRU_H + gx[u]];
            inn[u] = xr[2 * GRU_H + gx[u]];
        }
        // (2) prefetch next step's rows (2 points per 528B row)
        if (t + 1 < GRU_T) {
            int ttn = dir ? (GRU_T - 2 - t) : (t + 1);
            const float* pr = xp + ((long)ttn * GRU_B + b0 + lm) * NG2 + half * NG;
            __builtin_prefetch(pr, 0, 3);
        }

        // (3) A fragment from f16 h staging (two 16B LDS loads)
        v8h_t lo = *(const v8h_t*)(ha_lds + lm * 32 + 8 * half);
        v8h_t hi = *(const v8h_t*)(ha_lds + lm * 32 + 16 + 8 * half);
        v16h_t a;
#pragma unroll
        for (int e = 0; e < 8; ++e) { a[e] = lo[e]; a[8 + e] = hi[e]; }

        // (4) gh = h * w_hh^T over 5 column tiles
        v8f_t c[5];
#pragma unroll
        for (int j = 0; j < 5; ++j) {
            v8f_t cz = {};
            c[j] = __builtin_amdgcn_wmma_f32_16x16x32_f16(false, a, false, bw[j],
                                                          (short)0, cz, false, false);
        }

        // (5) scatter gh (+ b_hh) into [m][n] order
#pragma unroll
        for (int j = 0; j < 5; ++j) {
            int n = j * 16 + lm;
#pragma unroll
            for (int v = 0; v < 8; ++v)
                gh_lds[(v + 8 * half) * 80 + n] = c[j][v] + bb[j];
        }
        __syncthreads();

        // (6) GRU gate math: 352 outputs, 11 per lane, xp already in registers
#pragma unroll
        for (int u = 0; u < GATE_U; ++u) {
            int m = gm[u], hx = gx[u];
            float hr = gh_lds[m * 80 + hx];
            float hz = gh_lds[m * 80 + GRU_H + hx];
            float hn = gh_lds[m * 80 + 2 * GRU_H + hx];
            float hp = h_lds[m * 24 + hx];
            float r = 1.f / (1.f + __expf(-(ir[u] + hr)));
            float z = 1.f / (1.f + __expf(-(iz[u] + hz)));
            float n = tanhf(inn[u] + r * hn);
            float h = (1.f - z) * n + z * hp;
            h_lds[m * 24 + hx]  = h;
            ha_lds[m * 32 + hx] = (_Float16)h;
            hout[((long)tt * GRU_B + b0 + m) * KP1 + dir * GRU_H + hx] = (_Float16)h;
        }
        __syncthreads();
    }
}

// ---------------------------------------------------------------------------
// hs[b*22016 + t*22 + hx] = f16( h2[t,b,hx] + h2[t,b,22+hx] ); zero pad tail.
__global__ void sum_pack_kernel(const _Float16* __restrict__ h2,
                                _Float16* __restrict__ hs) {
    long i = (long)blockIdx.x * blockDim.x + threadIdx.x;
    const long total = (long)GRU_B * KPLIN;
    if (i >= total) return;
    int k = (int)(i % KPLIN);
    int b = (int)(i / KPLIN);
    float v = 0.f;
    if (k < KLIN) {
        int t  = k / GRU_H;
        int hx = k - t * GRU_H;
        const _Float16* p = h2 + ((long)t * GRU_B + b) * KP1;
        v = (float)p[hx] + (float)p[GRU_H + hx];
    }
    hs[i] = (_Float16)v;
}

// ---------------------------------------------------------------------------
// logits = y * w_cls^T + b_cls ; softmax over 4. One lane per batch row.
__global__ void head_kernel(const float* __restrict__ y,      // [256, 22]
                            const float* __restrict__ w_cls,  // [4, 22]
                            const float* __restrict__ b_cls,  // [4]
                            float* __restrict__ out)          // [256, 4]
{
    int b = blockIdx.x * blockDim.x + threadIdx.x;
    if (b >= GRU_B) return;
    float l[4];
#pragma unroll
    for (int cix = 0; cix < 4; ++cix) {
        float acc = b_cls[cix];
        for (int o = 0; o < GRU_H; ++o)
            acc += y[b * GRU_H + o] * w_cls[cix * GRU_H + o];
        l[cix] = acc;
    }
    float mx = fmaxf(fmaxf(l[0], l[1]), fmaxf(l[2], l[3]));
    float s = 0.f;
#pragma unroll
    for (int cix = 0; cix < 4; ++cix) { l[cix] = __expf(l[cix] - mx); s += l[cix]; }
    float inv = 1.f / s;
#pragma unroll
    for (int cix = 0; cix < 4; ++cix) out[b * 4 + cix] = l[cix] * inv;
}

// ---------------------------------------------------------------------------
extern "C" void kernel_launch(void* const* d_in, const int* in_sizes, int n_in,
                              void* d_out, int out_size, void* d_ws, size_t ws_size,
                              hipStream_t stream) {
    const float* x      = (const float*)d_in[0];
    const float* w_ih0  = (const float*)d_in[1];   // [2,66,22] -> [132,22]
    const float* w_hh0  = (const float*)d_in[2];   // [2,66,22]
    const float* b_ih0  = (const float*)d_in[3];   // [132]
    const float* b_hh0  = (const float*)d_in[4];
    const float* w_ih1  = (const float*)d_in[5];   // [2,66,44] -> [132,44]
    const float* w_hh1  = (const float*)d_in[6];
    const float* b_ih1  = (const float*)d_in[7];
    const float* b_hh1  = (const float*)d_in[8];
    const float* w_lin  = (const float*)d_in[9];   // [22, 22000]
    const float* b_lin  = (const float*)d_in[10];
    const float* w_cls  = (const float*)d_in[11];  // [4, 22]
    const float* b_cls  = (const float*)d_in[12];
    float* out = (float*)d_out;

    const long TB = (long)GRU_T * GRU_B;
    char* ws = (char*)d_ws;
    size_t off = 0;
    auto alloc = [&](size_t bytes) {
        char* p = ws + off;
        off += (bytes + 255) & ~(size_t)255;
        return p;
    };
    _Float16* xt_h  = (_Float16*)alloc(TB * KP0 * sizeof(_Float16));    // 16.4 MB
    float*    xp    = (float*)   alloc(TB * NG2 * sizeof(float));       // 135 MB (reused)
    _Float16* h1_h  = (_Float16*)alloc(TB * KP1 * sizeof(_Float16));    // 32.8 MB
    _Float16* h2_h  = (_Float16*)alloc(TB * KP1 * sizeof(_Float16));    // 32.8 MB
    _Float16* hs    = (_Float16*)alloc((size_t)GRU_B * KPLIN * sizeof(_Float16));
    float*    y     = (float*)   alloc((size_t)GRU_B * GRU_H * sizeof(float));
    _Float16* pk_ih0 = (_Float16*)alloc((size_t)9 * 1 * 512 * sizeof(_Float16));
    _Float16* pk_ih1 = (_Float16*)alloc((size_t)9 * 2 * 512 * sizeof(_Float16));
    _Float16* pk_hh0 = (_Float16*)alloc((size_t)2 * 5 * 512 * sizeof(_Float16));
    _Float16* pk_hh1 = (_Float16*)alloc((size_t)2 * 5 * 512 * sizeof(_Float16));
    _Float16* pk_lin = (_Float16*)alloc((size_t)2 * (KPLIN / 32) * 512 * sizeof(_Float16));

    const int KCLIN = KPLIN / 32;    // 688

    // 0. zero pad columns of h1/h2 (scan only writes cols 0..43 of each row)
    hipMemsetAsync(h1_h, 0, TB * KP1 * sizeof(_Float16), stream);
    hipMemsetAsync(h2_h, 0, TB * KP1 * sizeof(_Float16), stream);

    // 1. x -> f16 [T*B, 32] (transposed, zero-padded)
    {
        long total = TB * KP0;
        transpose_pad_kernel<<<dim3((unsigned)((total + 255) / 256)), 256, 0, stream>>>(x, xt_h);
    }
    // 2. pack all weights into B-fragment layouts
    {
        long t0 = 9L * 1 * 512;
        pack_w_kernel<<<dim3((unsigned)((t0 + 255) / 256)), 256, 0, stream>>>(
            w_ih0, NG2, GRU_H, 1, pk_ih0, t0);
        long t1 = 9L * 2 * 512;
        pack_w_kernel<<<dim3((unsigned)((t1 + 255) / 256)), 256, 0, stream>>>(
            w_ih1, NG2, 2 * GRU_H, 2, pk_ih1, t1);
        long th = 5L * 1 * 512;
        for (int dir = 0; dir < 2; ++dir) {
            pack_w_kernel<<<dim3((unsigned)((th + 255) / 256)), 256, 0, stream>>>(
                w_hh0 + (long)dir * NG * GRU_H, NG, GRU_H, 1, pk_hh0 + dir * 5 * 512, th);
            pack_w_kernel<<<dim3((unsigned)((th + 255) / 256)), 256, 0, stream>>>(
                w_hh1 + (long)dir * NG * GRU_H, NG, GRU_H, 1, pk_hh1 + dir * 5 * 512, th);
        }
        long tl = 2L * KCLIN * 512;
        pack_w_kernel<<<dim3((unsigned)((tl + 255) / 256)), 256, 0, stream>>>(
            w_lin, GRU_H, KLIN, KCLIN, pk_lin, tl);
    }
    // 3. layer-0 input projection: xp = xt * w_ih0^T + b_ih0  [T*B, 132]
    {
        dim3 grid((unsigned)(TB / (16 * GEMM_WAVES)), 9);
        wmma_gemm_f16<<<grid, 32 * GEMM_WAVES, 0, stream>>>(
            xt_h, KP0, (int)TB, 1, pk_ih0, b_ih0, NG2, xp, NG2, 0);
    }
    // 4. layer-0 bidirectional scan -> h1 [T*B, 64] f16
    gru_scan_kernel<<<dim3(GRU_B / 16, 2), 32, 0, stream>>>(xp, pk_hh0, b_hh0, h1_h);
    // 5. layer-1 input projection (K = 44 -> 64, KC = 2), reuse xp
    {
        dim3 grid((unsigned)(TB / (16 * GEMM_WAVES)), 9);
        wmma_gemm_f16<<<grid, 32 * GEMM_WAVES, 0, stream>>>(
            h1_h, KP1, (int)TB, 2, pk_ih1, b_ih1, NG2, xp, NG2, 0);
    }
    // 6. layer-1 bidirectional scan -> h2
    gru_scan_kernel<<<dim3(GRU_B / 16, 2), 32, 0, stream>>>(xp, pk_hh1, b_hh1, h2_h);
    // 7. enc = fwd + bwd, packed f16 [256, 22016]
    {
        long total = (long)GRU_B * KPLIN;
        sum_pack_kernel<<<dim3((unsigned)((total + 255) / 256)), 256, 0, stream>>>(h2_h, hs);
    }
    // 8. linear (+bias, ReLU): y[256,22] = relu(hs * w_lin^T + b_lin)
    {
        dim3 grid(GRU_B / (16 * GEMM_WAVES), 2);
        wmma_gemm_f16<<<grid, 32 * GEMM_WAVES, 0, stream>>>(
            hs, KPLIN, GRU_B, KCLIN, pk_lin, b_lin, GRU_H, y, GRU_H, 1);
    }
    // 9. classifier + softmax -> out [256,4]
    head_kernel<<<dim3(1), 256, 0, stream>>>(y, w_cls, b_cls, out);
}
